// LSTMModel_29532195127503
// MI455X (gfx1250) — matrix operand order repaired and checked by reading, not compiled, not run on hardware
//
#include <hip/hip_runtime.h>
#include <cmath>

// Problem constants (match reference): B=16, T=1024, F=512, L=4, 4F=2048
#define B_ 16
#define T_ 1024
#define F_ 512
#define L_ 4
#define G_ 2048
#define NWG_SCAN 16

typedef __attribute__((ext_vector_type(4)))  unsigned int v4u;
typedef __attribute__((ext_vector_type(8)))  float        v8f;
typedef __attribute__((ext_vector_type(16))) __bf16       v16bf;

union Frag16 { v4u u[2]; v16bf v; };

// ---------------- bf16 conversion (round-to-nearest-even) ----------------
__device__ __forceinline__ unsigned short f2bf(float x) {
  unsigned u = __float_as_uint(x);
  u += 0x7FFFu + ((u >> 16) & 1u);
  return (unsigned short)(u >> 16);
}

// ---------------- WMMA fragment loaders (row stride = F_ = 512) ----------------
// A (16x32 bf16, row-major): lane<16 holds row=lane, K in {kb..kb+7, kb+16..kb+23};
// lane>=16 holds row=lane-16, K in {kb+8..kb+15, kb+24..kb+31}.  (ISA 7.12.2)
__device__ __forceinline__ v16bf load_fragA(const unsigned short* base, int kbase, int lane) {
  const int r  = lane & 15;
  const int kh = (lane >> 4) & 1;
  const unsigned short* p = base + r * F_ + kbase + kh * 8;
  Frag16 f;
  f.u[0] = *reinterpret_cast<const v4u*>(p);
  f.u[1] = *reinterpret_cast<const v4u*>(p + 16);
  return f.v;
}
// B (32x16 bf16) sourced from B^T rows (row-major [N x K]): lane<16 -> col=lane,
// K in {kb..kb+7, kb+16..kb+23}; lane>=16 -> col=lane-16, K in {kb+8..kb+15, kb+24..kb+31}.
__device__ __forceinline__ v16bf load_fragB(const unsigned short* base, int kbase, int lane) {
  const int n  = lane & 15;
  const int kh = (lane >> 4) & 1;
  const unsigned short* p = base + n * F_ + kbase + kh * 8;
  Frag16 f;
  f.u[0] = *reinterpret_cast<const v4u*>(p);
  f.u[1] = *reinterpret_cast<const v4u*>(p + 16);
  return f.v;
}

__device__ __forceinline__ v8f wmma_bf16(v16bf a, v16bf b, v8f c) {
  // (neg_a, A, neg_b, B, c_mod, C, reuse_a, reuse_b)
  return __builtin_amdgcn_wmma_f32_16x16x32_bf16(false, a, false, b, (short)0, c, false, false);
}

// ---------------- Tensor Data Mover: 2D bf16 tile -> LDS ----------------
#if __has_builtin(__builtin_amdgcn_tensor_load_to_lds)
#define USE_TDM 1
typedef __attribute__((ext_vector_type(4))) unsigned int tdmv4u;
typedef __attribute__((ext_vector_type(8))) int          tdmv8i;
typedef __attribute__((ext_vector_type(4))) int          tdmv4i;

__device__ __forceinline__ void tdm_load_2d_bf16(unsigned lds_off, const void* gptr,
                                                 unsigned rows, unsigned cols,
                                                 unsigned row_stride_elems) {
  unsigned long long ga = (unsigned long long)(uintptr_t)gptr;
  tdmv4u g0;
  g0[0] = 1u;                                             // count=1 valid user descriptor
  g0[1] = lds_off;                                        // lds_addr
  g0[2] = (unsigned)(ga & 0xFFFFFFFFu);                   // global_addr[31:0]
  g0[3] = (unsigned)((ga >> 32) & 0x01FFFFFFu) | (2u << 30); // global_addr[56:32] | type=2
  tdmv8i g1;
  const unsigned data_size = 1u;                          // 2-byte elements
  g1[0] = (int)(data_size << 16);                         // wg_mask=0 | data_size
  g1[1] = (int)((cols & 0xFFFFu) << 16);                  // tensor_dim0[15:0] at bits[63:48]
  g1[2] = (int)(((cols >> 16) & 0xFFFFu) | ((rows & 0xFFFFu) << 16)); // dim0 hi | dim1 lo
  g1[3] = (int)(((rows >> 16) & 0xFFFFu) | ((cols & 0xFFFFu) << 16)); // dim1 hi | tile_dim0
  g1[4] = (int)(rows & 0xFFFFu);                          // tile_dim1 | tile_dim2=0
  g1[5] = (int)row_stride_elems;                          // tensor_dim0_stride[31:0]
  g1[6] = 0;                                              // stride0 hi | stride1 lo
  g1[7] = 0;                                              // stride1 hi
  tdmv4i z4 = {0, 0, 0, 0};
#if defined(__clang_major__) && __clang_major__ >= 23
  tdmv8i z8 = {0, 0, 0, 0, 0, 0, 0, 0};
  __builtin_amdgcn_tensor_load_to_lds(g0, g1, z4, z4, z8, 0);
#else
  __builtin_amdgcn_tensor_load_to_lds(g0, g1, z4, z4, 0);
#endif
}
#else
#define USE_TDM 0
#endif

// ---------------- device-scope barrier across the 16 scan workgroups ----------------
__device__ __forceinline__ void gbar(unsigned* cnt, unsigned* phase, unsigned nwg) {
  __threadfence();            // make h stores visible at agent scope
  __syncthreads();
  if (threadIdx.x == 0) {
    unsigned ph  = __hip_atomic_load(phase, __ATOMIC_ACQUIRE, __HIP_MEMORY_SCOPE_AGENT);
    unsigned arr = __hip_atomic_fetch_add(cnt, 1u, __ATOMIC_ACQ_REL, __HIP_MEMORY_SCOPE_AGENT);
    if (arr == nwg - 1u) {
      __hip_atomic_store(cnt, 0u, __ATOMIC_RELAXED, __HIP_MEMORY_SCOPE_AGENT);
      __hip_atomic_fetch_add(phase, 1u, __ATOMIC_ACQ_REL, __HIP_MEMORY_SCOPE_AGENT);
    } else {
      while (__hip_atomic_load(phase, __ATOMIC_ACQUIRE, __HIP_MEMORY_SCOPE_AGENT) == ph) {
        __builtin_amdgcn_s_sleep(1);
      }
    }
  }
  __syncthreads();
}

// ====================== prep kernels ======================
// x [B,T,F] f32 -> seq [T,B,F] bf16
__global__ void k_convert_x(const float* __restrict__ x, unsigned short* __restrict__ out) {
  int i = blockIdx.x * blockDim.x + threadIdx.x;
  const int total = T_ * B_ * F_;
  if (i >= total) return;
  int k = i % F_;
  int b = (i / F_) % B_;
  int t = i / (F_ * B_);
  out[i] = f2bf(x[((size_t)b * T_ + t) * F_ + k]);
}

// Wi/Wh [L,F,4F] f32 -> WiT/WhT [L,4F,F] bf16 (B^T layout for fragment loads)
__global__ void k_transpose_w(const float* __restrict__ Wi, const float* __restrict__ Wh,
                              unsigned short* __restrict__ WiT, unsigned short* __restrict__ WhT) {
  int i = blockIdx.x * blockDim.x + threadIdx.x;
  const int total = L_ * F_ * G_;
  if (i >= total) return;
  int g = i % G_;
  int k = (i / G_) % F_;
  int l = i / (G_ * F_);
  size_t dst = ((size_t)l * G_ + g) * F_ + k;
  WiT[dst] = f2bf(Wi[i]);
  WhT[dst] = f2bf(Wh[i]);
}

__global__ void k_init_bar(unsigned* __restrict__ bar) {
  if (threadIdx.x < 16) bar[threadIdx.x] = 0u;
}

// ====================== phase 1: xg = x @ Wi + b  (WMMA GEMM) ======================
// Xin [T*16, 512] bf16 (t-major rows), WiT_l [2048, 512] bf16, XG [T*16, 2048] f32
__global__ __launch_bounds__(256) void xg_gemm(const unsigned short* __restrict__ Xin,
                                               const unsigned short* __restrict__ WiT_l,
                                               const float* __restrict__ bias,
                                               float* __restrict__ XG) {
  const int tid  = threadIdx.x;
  const int lane = tid & 31;
  const int wave = tid >> 5;
  const int t      = blockIdx.x >> 4;        // M-tile == timestep (16 batch rows)
  const int ngroup = blockIdx.x & 15;
  const int ntile  = ngroup * 8 + wave;      // 128 N-tiles over 2048 gate columns

  const unsigned short* Atile = Xin   + (size_t)t * 16 * F_;
  const unsigned short* Btile = WiT_l + (size_t)ntile * 16 * F_;

  v8f acc;
#pragma unroll
  for (int r = 0; r < 8; ++r) acc[r] = 0.0f;

#pragma unroll
  for (int kk = 0; kk < F_; kk += 32) {
    v16bf a = load_fragA(Atile, kk, lane);
    v16bf b = load_fragB(Btile, kk, lane);
    acc = wmma_bf16(a, b, acc);
  }

  const int col  = ntile * 16 + (lane & 15);
  const int row0 = (lane >> 4) * 8;
  const float bv = bias[col];
  float* outp = XG + (size_t)t * 16 * G_ + col;
#pragma unroll
  for (int r = 0; r < 8; ++r) outp[(size_t)(row0 + r) * G_] = acc[r] + bv;
}

// ====================== phase 2: recurrent scan (persistent, 16 WGPs) ======================
// Each workgroup g owns hidden units j in [g*32, g*32+32) -> gate cols {j, 512+j, 1024+j, 1536+j}.
// Wh slice (128 rows x 512 K, bf16) lives in LDS for the whole layer (TDM-loaded);
// h_{t-1} is staged into LDS once per step, so the GEMM runs entirely out of LDS.
__global__ __launch_bounds__(256) void lstm_scan(const unsigned short* __restrict__ WhT_l, // [2048,512] bf16
                                                 const float* __restrict__ XG,             // [T*16,2048] f32
                                                 unsigned short* __restrict__ Yout,        // [T*16,512] bf16
                                                 float* __restrict__ Dout,                 // [B,T,F] f32 or null
                                                 unsigned short* __restrict__ HBUF,        // [2][16][512] bf16
                                                 unsigned* __restrict__ BAR) {
  __shared__ __align__(16) unsigned short sWh[128 * F_];   // 128 KiB: resident Wh slice
  __shared__ __align__(16) unsigned short sH[16 * F_];     // 16 KiB: h_{t-1} stage
  __shared__ __align__(16) float sGates[16 * 128];         // 8 KiB: gate tile staging
  __shared__ __align__(16) float sC[16 * 32];              // 2 KiB: cell state slice

  const int g    = blockIdx.x;
  const int tid  = threadIdx.x;
  const int lane = tid & 31;
  const int wave = tid >> 5;

  // ---- stage Wh slice into LDS via TDM: 4 gate blocks of 32 rows x 512 ----
#if USE_TDM
  if (wave == 0) {
#pragma unroll
    for (int q = 0; q < 4; ++q) {
      const unsigned short* gsrc = WhT_l + (size_t)(q * F_ + g * 32) * F_;
      unsigned lds_off = (unsigned)(uintptr_t)(&sWh[q * 32 * F_]);
      tdm_load_2d_bf16(lds_off, gsrc, /*rows=*/32, /*cols=*/F_, /*stride=*/F_);
    }
    __builtin_amdgcn_s_wait_tensorcnt(0);
  }
#else
  for (int i = tid; i < (128 * F_) / 8; i += 256) {
    int lr = (i * 8) >> 9;          // local row 0..127
    int kc = (i * 8) & (F_ - 1);
    int grow = (lr >> 5) * F_ + g * 32 + (lr & 31);
    reinterpret_cast<v4u*>(sWh)[i] =
        *reinterpret_cast<const v4u*>(WhT_l + (size_t)grow * F_ + kc);
  }
#endif

  // ---- zero h double-buffer slice and cell state ----
  for (int i = tid; i < 2 * 16 * 32; i += 256) {
    int buf = i >> 9, rem = i & 511, b = rem >> 5, u = rem & 31;
    HBUF[buf * 16 * F_ + b * F_ + g * 32 + u] = 0;
  }
  for (int i = tid; i < 16 * 32; i += 256) sC[i] = 0.0f;
  __syncthreads();
  gbar(BAR, BAR + 1, NWG_SCAN);

  // wave-local geometry: 8 waves x one 16-col N-tile = 128 gate cols
  const int lc    = wave * 16 + (lane & 15);     // local gate column 0..127
  const int gateq = lc >> 5;                     // 0=i 1=f 2=g 3=o
  const int uq    = lc & 31;                     // hidden unit within slice
  const int gcol  = gateq * F_ + g * 32 + uq;    // global gate column
  const int row0  = (lane >> 4) * 8;
  const unsigned short* Bt = sWh + (size_t)wave * 16 * F_;

  for (int t = 0; t < T_; ++t) {
    const unsigned short* hread = HBUF + (size_t)(t & 1) * 16 * F_;
    unsigned short* hwrite      = HBUF + (size_t)((t + 1) & 1) * 16 * F_;

    // ---- stage h_{t-1} (16x512 bf16 = 16 KB) into LDS, four b128 per lane ----
    {
      const v4u* src = reinterpret_cast<const v4u*>(hread);
      v4u*       dst = reinterpret_cast<v4u*>(sH);
#pragma unroll
      for (int i = 0; i < 4; ++i) dst[tid + i * 256] = src[tid + i * 256];
    }

    // prefetch next timestep's xg tile while this step computes
    if (t + 1 < T_) {
      const float* nxt = XG + (size_t)(t + 1) * 16 * G_ + (size_t)row0 * G_ + gcol;
      __builtin_prefetch(nxt, 0, 1);
    }

    // acc = xg[t] (bias folded in phase 1)
    const float* xg = XG + (size_t)t * 16 * G_;
    v8f acc;
#pragma unroll
    for (int r = 0; r < 8; ++r) acc[r] = xg[(size_t)(row0 + r) * G_ + gcol];

    __syncthreads();   // sH ready

    // gates += h_{t-1} @ Wh_slice  (A and B both from LDS)
#pragma unroll
    for (int kk = 0; kk < F_; kk += 32) {
      v16bf a = load_fragA(sH, kk, lane);
      v16bf b = load_fragB(Bt, kk, lane);
      acc = wmma_bf16(a, b, acc);
    }

#pragma unroll
    for (int r = 0; r < 8; ++r) sGates[(row0 + r) * 128 + lc] = acc[r];
    __syncthreads();

    // elementwise LSTM cell update: 16 batch x 32 units
    for (int i = tid; i < 16 * 32; i += 256) {
      int b = i >> 5, u = i & 31;
      float gi = sGates[b * 128 + u];
      float gf = sGates[b * 128 + 32 + u];
      float gg = sGates[b * 128 + 64 + u];
      float go = sGates[b * 128 + 96 + u];
      float si = 1.0f / (1.0f + __expf(-gi));
      float sf = 1.0f / (1.0f + __expf(-gf));
      float so = 1.0f / (1.0f + __expf(-go));
      float nc = sf * sC[i] + si * tanhf(gg);
      float nh = so * tanhf(nc);
      sC[i] = nc;
      int j = g * 32 + u;
      unsigned short nhb = f2bf(nh);
      hwrite[b * F_ + j] = nhb;
      Yout[(size_t)(t * 16 + b) * F_ + j] = nhb;
      if (Dout) Dout[((size_t)b * T_ + t) * F_ + j] = nh;
    }

    __builtin_amdgcn_s_cluster_barrier();   // HW cluster path if dispatched clustered (NOP else)
    gbar(BAR, BAR + 1, NWG_SCAN);           // portable device-scope step barrier
  }
}

// ====================== host launcher ======================
extern "C" void kernel_launch(void* const* d_in, const int* in_sizes, int n_in,
                              void* d_out, int out_size, void* d_ws, size_t ws_size,
                              hipStream_t stream) {
  (void)in_sizes; (void)n_in; (void)out_size; (void)ws_size;
  const float* x  = (const float*)d_in[0];
  const float* Wi = (const float*)d_in[1];
  const float* Wh = (const float*)d_in[2];
  const float* bs = (const float*)d_in[3];

  char* ws = (char*)d_ws;
  size_t ofs = 0;
  auto suballoc = [&](size_t bytes) -> void* {
    void* p = ws + ofs;
    ofs = (ofs + bytes + 255) & ~(size_t)255;
    return p;
  };
  unsigned short* seq0 = (unsigned short*)suballoc((size_t)T_ * B_ * F_ * 2);   // 16 MiB
  unsigned short* seq1 = (unsigned short*)suballoc((size_t)T_ * B_ * F_ * 2);   // 16 MiB
  unsigned short* WiT  = (unsigned short*)suballoc((size_t)L_ * G_ * F_ * 2);   // 8 MiB
  unsigned short* WhT  = (unsigned short*)suballoc((size_t)L_ * G_ * F_ * 2);   // 8 MiB
  float*          XG   = (float*)         suballoc((size_t)T_ * B_ * G_ * 4);   // 128 MiB
  unsigned short* HBUF = (unsigned short*)suballoc((size_t)2 * B_ * F_ * 2);    // 32 KiB
  unsigned*       BAR  = (unsigned*)      suballoc(64);

  {
    int n = T_ * B_ * F_;
    k_convert_x<<<(n + 255) / 256, 256, 0, stream>>>(x, seq0);
  }
  {
    int n = L_ * F_ * G_;
    k_transpose_w<<<(n + 255) / 256, 256, 0, stream>>>(Wi, Wh, WiT, WhT);
  }
  k_init_bar<<<1, 64, 0, stream>>>(BAR);

  unsigned short* xin  = seq0;
  unsigned short* yout = seq1;
  for (int l = 0; l < L_; ++l) {
    xg_gemm<<<dim3(T_ * 16), 256, 0, stream>>>(xin, WiT + (size_t)l * G_ * F_,
                                               bs + (size_t)l * G_, XG);
    lstm_scan<<<dim3(NWG_SCAN), 256, 0, stream>>>(WhT + (size_t)l * G_ * F_, XG, yout,
                                                  (l == L_ - 1) ? (float*)d_out : nullptr,
                                                  HBUF, BAR);
    unsigned short* tmp = xin; xin = yout; yout = tmp;
  }
}
